// PointnetSAModuleMSG_37237366456768
// MI455X (gfx1250) — matrix operand order repaired and checked by reading, not compiled, not run on hardware
//
#include <hip/hip_runtime.h>
#include <stdint.h>

typedef __attribute__((ext_vector_type(16))) __bf16          v16bf;
typedef __attribute__((ext_vector_type(16))) unsigned short  v16u;
typedef __attribute__((ext_vector_type(8)))  float           v8f;
typedef __attribute__((ext_vector_type(4)))  unsigned int    u32x4;
typedef __attribute__((ext_vector_type(8)))  int             i32x8;
typedef __attribute__((ext_vector_type(4)))  int             i32x4;

#define N_PTS   4096
#define BATCH   4
#define BN_PTS  (BATCH * N_PTS)   // 16384
#define CFEAT   128
#define C_IN    131
#define C_INP   160               // padded K for phi (5 tiles of 32)
#define C_OUT   256
#define C_SM    64
#define NS0     32
#define NS1     64
#define R0SQ    0.04f
#define R1SQ    0.16f

#define APHI_BYTES_PER_SC  81920u   // 16 ot * 5 kt * 32 lanes * 32 B

// ---- workspace byte offsets ----
#define OFF_IDX0  0x000000u   // int[16384*32]            = 2 MB
#define OFF_IDX1  0x200000u   // int[16384*64]            = 4 MB
#define OFF_APHI  0x600000u   // ushort 2*16*5*32*16      = 160 KB
#define OFF_APSI  0x630000u   // ushort 2*4*8*32*16       = 64 KB
#define OFF_S1T1  0x640000u   // float 2*(160+160)
#define OFF_AB    0x642000u   // float 2*(256+256)
#define OFF_BPSI  0x644000u   // float 64
#define OFF_MV    0x650000u   // ushort 2*256*16384       = 16 MB

struct Params {
    const float* bn_cin_g[2]; const float* bn_cin_b[2];
    const float* bn_cin_m[2]; const float* bn_cin_v[2];
    const float* w_phi[2];    const float* b_phi[2];
    const float* bn_phi_g[2]; const float* bn_phi_b[2];
    const float* bn_phi_m[2]; const float* bn_phi_v[2];
    const float* w_psi[2];    const float* b_psi[2];
};

__device__ __forceinline__ unsigned short f2bf(float f) {
    unsigned u = __float_as_uint(f);
    unsigned r = u + 0x7FFFu + ((u >> 16) & 1u);   // round-to-nearest-even
    return (unsigned short)(r >> 16);
}

__device__ __forceinline__ v8f wmma_bf16(v16u a, v16u b, v8f c) {
    return __builtin_amdgcn_wmma_f32_16x16x32_bf16(
        false, __builtin_bit_cast(v16bf, a),
        false, __builtin_bit_cast(v16bf, b),
        (short)0, c, false, false);
}

struct AFrag2 { u32x4 lo, hi; };   // 32B A fragment as two 16B halves

// ---------------------------------------------------------------------------
// out[0:49152) = xyz ; out channels 0..127 = 2*features
// ---------------------------------------------------------------------------
__global__ void k_init_out(const float* __restrict__ xyz,
                           const float* __restrict__ feat,
                           float* __restrict__ dout) {
    int i = blockIdx.x * 256 + threadIdx.x;
    if (i < 3 * BN_PTS) { dout[i] = xyz[i]; return; }
    int j = i - 3 * BN_PTS;
    if (j < BATCH * CFEAT * N_PTS) {
        int b = j >> 19;              // / (128*4096)
        int r = j & 524287;           // c*4096 + s, c < 128
        dout[3 * BN_PTS + (size_t)b * 192 * N_PTS + r] = 2.0f * feat[j];
    }
}

// ---------------------------------------------------------------------------
// Ball query: one wave per query point, ordered streaming compaction.
// ---------------------------------------------------------------------------
__global__ void __launch_bounds__(128) k_ballquery(const float* __restrict__ xyz,
                                                   int* __restrict__ idx0,
                                                   int* __restrict__ idx1) {
    int wave = threadIdx.x >> 5, lane = threadIdx.x & 31;
    int g = blockIdx.x * 4 + wave;            // 0..16383
    int b = g >> 12, s = g & 4095;
    const float* xb = xyz + (size_t)b * N_PTS * 3;
    float qx = xb[s * 3 + 0], qy = xb[s * 3 + 1], qz = xb[s * 3 + 2];
    int cnt0 = 0, cnt1 = 0;
    unsigned lt = (1u << lane) - 1u;
    for (int c0 = 0; c0 < N_PTS; c0 += 32) {
        int p = c0 + lane;
        float dx = xb[p * 3 + 0] - qx;
        float dy = xb[p * 3 + 1] - qy;
        float dz = xb[p * 3 + 2] - qz;
        float d2 = dx * dx + dy * dy + dz * dz;
        bool w0 = d2 < R0SQ, w1 = d2 < R1SQ;
        unsigned m0 = (unsigned)__ballot(w0);
        unsigned m1 = (unsigned)__ballot(w1);
        if (cnt0 < NS0 && w0) {
            int k = cnt0 + __popc(m0 & lt);
            if (k < NS0) idx0[(size_t)g * NS0 + k] = p;
        }
        if (cnt1 < NS1 && w1) {
            int k = cnt1 + __popc(m1 & lt);
            if (k < NS1) idx1[(size_t)g * NS1 + k] = p;
        }
        cnt0 += __popc(m0); cnt1 += __popc(m1);
        if (cnt0 >= NS0 && cnt1 >= NS1) break;
    }
    // self point (d2=0) guarantees at least one hit; fill tail with first index
    int f0 = idx0[(size_t)g * NS0];
    int f1 = idx1[(size_t)g * NS1];
    int c0f = cnt0 < NS0 ? cnt0 : NS0;
    int c1f = cnt1 < NS1 ? cnt1 : NS1;
    for (int k = c0f + lane; k < NS0; k += 32) idx0[(size_t)g * NS0 + k] = f0;
    for (int k = c1f + lane; k < NS1; k += 32) idx1[(size_t)g * NS1 + k] = f1;
}

// ---------------------------------------------------------------------------
// Fold BN into per-channel scale/bias: pre-GEMM (s1,t1), post-GEMM (alpha,beta)
// ---------------------------------------------------------------------------
__global__ void k_prep_scalars(Params P, char* ws) {
    float* s1t1 = (float*)(ws + OFF_S1T1);
    float* ab   = (float*)(ws + OFF_AB);
    float* bp   = (float*)(ws + OFF_BPSI);
    int t = threadIdx.x;
    if (t < 2 * C_INP) {
        int sc = t / C_INP, c = t % C_INP;
        float s = 0.f, tt = 0.f;
        if (c < C_IN) {
            float gg = P.bn_cin_g[sc][c], be = P.bn_cin_b[sc][c];
            float mm = P.bn_cin_m[sc][c], vv = P.bn_cin_v[sc][c];
            s  = gg * rsqrtf(vv + 1e-5f);
            tt = be - mm * s;
        }
        s1t1[sc * 2 * C_INP + c]         = s;
        s1t1[sc * 2 * C_INP + C_INP + c] = tt;
    } else if (t < 2 * C_INP + 2 * C_OUT) {
        int j = t - 2 * C_INP; int sc = j / C_OUT, m = j % C_OUT;
        float gg = P.bn_phi_g[sc][m], be = P.bn_phi_b[sc][m];
        float mm = P.bn_phi_m[sc][m], vv = P.bn_phi_v[sc][m];
        float a = gg * rsqrtf(vv + 1e-5f);
        ab[sc * 2 * C_OUT + m]          = a;
        ab[sc * 2 * C_OUT + C_OUT + m]  = be - mm * a + P.b_phi[sc][m] * a;
    } else if (t < 2 * C_INP + 2 * C_OUT + C_SM) {
        int o = t - (2 * C_INP + 2 * C_OUT);
        bp[o] = P.b_psi[0][o] + P.b_psi[1][o];
    }
}

// A-fragment K offsets: contiguous order, k = 16*(lane/16) + e (matches B fills)
__device__ __forceinline__ int a_koff(int lane, int e) {
    return ((lane < 16) ? 0 : 16) + e;
}

// Swizzle w_phi (256x131, pad K->160) into A fragments: [sc][ot:16][kt:5][lane][16]
__global__ void k_prep_Aphi(Params P, char* ws) {
    int gid = blockIdx.x * 256 + threadIdx.x;
    if (gid >= 2 * 16 * 5 * 32) return;
    int lane = gid & 31; int q = gid >> 5;
    int kt = q % 5; int ot = (q / 5) & 15; int sc = q / 80;
    const float* W = P.w_phi[sc];
    int m = ot * 16 + (lane & 15);
    unsigned short* dst = (unsigned short*)(ws + OFF_APHI) + (size_t)gid * 16;
    #pragma unroll
    for (int e = 0; e < 16; ++e) {
        int c = kt * 32 + a_koff(lane, e);
        float w = (c < C_IN) ? W[(size_t)m * C_IN + c] : 0.f;
        dst[e] = f2bf(w);
    }
}

// Swizzle w_psi (64x256) into A fragments: [sc][ot:4][kt:8][lane][16]
__global__ void k_prep_Apsi(Params P, char* ws) {
    int gid = blockIdx.x * 256 + threadIdx.x;
    if (gid >= 2 * 4 * 8 * 32) return;
    int lane = gid & 31; int q = gid >> 5;
    int kt = q % 8; int ot = (q / 8) & 3; int sc = q / 32;
    const float* W = P.w_psi[sc];
    int m = ot * 16 + (lane & 15);
    unsigned short* dst = (unsigned short*)(ws + OFF_APSI) + (size_t)gid * 16;
    #pragma unroll
    for (int e = 0; e < 16; ++e) {
        int c = kt * 32 + a_koff(lane, e);
        dst[e] = f2bf(W[(size_t)m * C_OUT + c]);
    }
}

// ---------------------------------------------------------------------------
// TDM: stage one scale's w_phi A-fragments (80 KB, contiguous) into LDS.
// Minimal 1D descriptor per ISA ch.8: count=1, type=2(image), data_size=8B,
// tile_dim0 = 10240 elements, tile_dim1 = 0 (1D), stride = tensor_dim0.
// 6-arg builtin (clang-23 / therock signature): groups 2,3 and the extra
// int32x8 group are zero (unused for a 1D tile).
// ---------------------------------------------------------------------------
__device__ __forceinline__ void tdm_stage_A(char* smem, const char* gsrc) {
#if __has_builtin(__builtin_amdgcn_tensor_load_to_lds)
    if (threadIdx.x == 0) {
        unsigned long long ga = (unsigned long long)(uintptr_t)gsrc;
        unsigned lds_addr = (unsigned)(uintptr_t)smem;   // LDS aperture: low 32b
        const unsigned elems = APHI_BYTES_PER_SC / 8u;   // 10240 x 8B
        u32x4 g0;
        g0[0] = 1u;                                       // count=1 (valid D#)
        g0[1] = lds_addr;                                 // lds_addr
        g0[2] = (unsigned)ga;                             // global_addr[31:0]
        g0[3] = (unsigned)((ga >> 32) & 0x1FFFFFFu) | (2u << 30); // addr[56:32] | type=2
        i32x8 g1;
        g1[0] = (int)(3u << 16);                          // data_size=8B, wg_mask=0
        g1[1] = (int)((elems & 0xFFFFu) << 16);           // tensor_dim0[15:0]
        g1[2] = (int)(((elems >> 16) & 0xFFFFu) | (1u << 16)); // td0[31:16] | tensor_dim1=1
        g1[3] = (int)((elems & 0xFFFFu) << 16);           // td1 hi=0 | tile_dim0
        g1[4] = 0;                                        // tile_dim1=0 (1D), tile_dim2=0
        g1[5] = (int)elems;                               // tensor_dim0_stride lo
        g1[6] = 0;                                        // stride hi | dim1_stride lo
        g1[7] = 0;
        i32x4 z4 = {0, 0, 0, 0};
        i32x8 z8 = {0, 0, 0, 0, 0, 0, 0, 0};
        __builtin_amdgcn_tensor_load_to_lds(g0, g1, z4, z4, z8, 0);
        __builtin_amdgcn_s_wait_tensorcnt(0);
    }
#else
    // fallback: cooperative 16B copies
    const u32x4* src = (const u32x4*)gsrc;
    u32x4* dst = (u32x4*)smem;
    for (int i = threadIdx.x; i < (int)(APHI_BYTES_PER_SC / 16u); i += 256)
        dst[i] = src[i];
#endif
}

// ---------------------------------------------------------------------------
// Main fused kernel: 256 threads = 8 waves, one point per wave.
// w_phi A-fragments staged in LDS by the Tensor Data Mover; B fragments built
// by gather+BN+ReLU->bf16; 16x16x32 bf16 WMMA; post-BN+ReLU; sample-max via
// ds_max_num_f32; 256-wide max vector written to ws (bf16) for the psi GEMM.
// Dynamic LDS: [0,81920) A fragments, [81920,+8KB) per-wave smax.
// ---------------------------------------------------------------------------
__global__ void __launch_bounds__(256) k_phi(const float* __restrict__ xyz,
                                             const float* __restrict__ feat,
                                             char* ws) {
    extern __shared__ char smem[];
    AFrag2* ldsA = (AFrag2*)smem;                       // 2560 fragments
    float*  smax = (float*)(smem + APHI_BYTES_PER_SC);  // 8 * 256 floats
    int wave = threadIdx.x >> 5, lane = threadIdx.x & 31;
    int g = blockIdx.x * 8 + wave;
    int b = g >> 12, s = g & 4095;
    int half  = lane & 15;
    int kbase = (lane < 16) ? 0 : 16;
    int mbase = (lane < 16) ? 0 : 8;
    const float* xb = xyz + (size_t)b * N_PTS * 3;
    float qx = xb[s * 3 + 0], qy = xb[s * 3 + 1], qz = xb[s * 3 + 2];
    unsigned short* mv = (unsigned short*)(ws + OFF_MV);
    const int* idx0 = (const int*)(ws + OFF_IDX0) + (size_t)g * NS0;
    const int* idx1 = (const int*)(ws + OFF_IDX1) + (size_t)g * NS1;
    float* smaxw = smax + wave * C_OUT;

    for (int sc = 0; sc < 2; ++sc) {
        __syncthreads();   // previous scale's A reads complete before overwrite
        tdm_stage_A(smem, (const char*)ws + OFF_APHI + (unsigned)sc * APHI_BYTES_PER_SC);
        __syncthreads();   // staged A visible to all waves

        const float* s1 = (const float*)(ws + OFF_S1T1) + sc * 2 * C_INP;
        const float* t1 = s1 + C_INP;
        const float* al = (const float*)(ws + OFF_AB) + sc * 2 * C_OUT;
        const float* be = al + C_OUT;
        const int* idx = sc ? idx1 : idx0;
        int nct = sc ? (NS1 / 16) : (NS0 / 16);

        for (int i = lane; i < C_OUT; i += 32)
            __hip_atomic_store(&smaxw[i], 0.f, __ATOMIC_RELAXED,
                               __HIP_MEMORY_SCOPE_WAVEFRONT);

        for (int ct = 0; ct < nct; ++ct) {
            int p = idx[ct * 16 + half];                 // column sample
            const float* fp = feat + (size_t)b * CFEAT * N_PTS + p;
            float px = xb[p * 3 + 0] - qx;
            float py = xb[p * 3 + 1] - qy;
            float pz = xb[p * 3 + 2] - qz;

            v16u Bf[5];
            #pragma unroll
            for (int kt = 0; kt < 5; ++kt) {
                v16u bb;
                #pragma unroll
                for (int e = 0; e < 16; ++e) {
                    int c = kt * 32 + kbase + e;
                    float v = 0.f;
                    if (c < C_IN) {
                        float raw;
                        if (c == 0) raw = px;
                        else if (c == 1) raw = py;
                        else if (c == 2) raw = pz;
                        else raw = fp[(size_t)(c - 3) * N_PTS];
                        v = fmaxf(raw * s1[c] + t1[c], 0.f);   // BN + ReLU
                    }
                    bb[e] = f2bf(v);
                }
                Bf[kt] = bb;
            }

            for (int ot = 0; ot < 16; ++ot) {
                v8f acc = {0.f, 0.f, 0.f, 0.f, 0.f, 0.f, 0.f, 0.f};
                #pragma unroll
                for (int kt = 0; kt < 5; ++kt) {
                    AFrag2 af = ldsA[(ot * 5 + kt) * 32 + lane];   // ds_load x2
                    acc = wmma_bf16(__builtin_bit_cast(v16u, af), Bf[kt], acc);
                }
                #pragma unroll
                for (int v = 0; v < 8; ++v) {
                    int m = ot * 16 + mbase + v;
                    float z = fmaxf(acc[v] * al[m] + be[m], 0.f);  // BN + ReLU
                    __hip_atomic_fetch_max(&smaxw[m], z, __ATOMIC_RELAXED,
                                           __HIP_MEMORY_SCOPE_WAVEFRONT);
                }
            }
        }
        // write max vector (bf16), column-major per point for the psi GEMM
        for (int i = lane; i < C_OUT; i += 32) {
            float mval = __hip_atomic_load(&smaxw[i], __ATOMIC_RELAXED,
                                           __HIP_MEMORY_SCOPE_WAVEFRONT);
            mv[(size_t)sc * C_OUT * BN_PTS + (size_t)i * BN_PTS + g] = f2bf(mval);
        }
    }
}

// ---------------------------------------------------------------------------
// psi: (64x256)x(256x16384) WMMA GEMM, both scales summed in one accumulator.
// ---------------------------------------------------------------------------
__global__ void __launch_bounds__(128) k_psi(const char* __restrict__ ws,
                                             float* __restrict__ dout) {
    int wave = threadIdx.x >> 5, lane = threadIdx.x & 31;
    int tile = blockIdx.x * 4 + wave;     // 0..1023
    int half  = lane & 15;
    int kbase = (lane < 16) ? 0 : 16;
    int mbase = (lane < 16) ? 0 : 8;
    int col = tile * 16 + half;           // global point index
    const v16u* Af = (const v16u*)(ws + OFF_APSI);
    const unsigned short* mv = (const unsigned short*)(ws + OFF_MV);
    const float* bp = (const float*)(ws + OFF_BPSI);
    int b = col >> 12, s = col & 4095;
    float* outb = dout + 3 * BN_PTS + (size_t)b * 192 * N_PTS + s;

    for (int ot = 0; ot < 4; ++ot) {
        v8f acc = {0.f, 0.f, 0.f, 0.f, 0.f, 0.f, 0.f, 0.f};
        for (int sc = 0; sc < 2; ++sc) {
            #pragma unroll
            for (int kt = 0; kt < 8; ++kt) {
                v16u bb;
                #pragma unroll
                for (int e = 0; e < 16; ++e) {
                    int c = kt * 32 + kbase + e;
                    bb[e] = mv[(size_t)sc * C_OUT * BN_PTS + (size_t)c * BN_PTS + col];
                }
                v16u a = Af[(size_t)((sc * 4 + ot) * 8 + kt) * 32 + lane];
                acc = wmma_bf16(a, bb, acc);
            }
        }
        #pragma unroll
        for (int v = 0; v < 8; ++v) {
            int o = ot * 16 + mbase + v;
            outb[(size_t)(128 + o) * N_PTS] = acc[v] + bp[o];
        }
    }
}

// ---------------------------------------------------------------------------
extern "C" void kernel_launch(void* const* d_in, const int* in_sizes, int n_in,
                              void* d_out, int out_size, void* d_ws, size_t ws_size,
                              hipStream_t stream) {
    const float* xyz  = (const float*)d_in[0];
    const float* feat = (const float*)d_in[1];
    float* dout = (float*)d_out;
    char*  ws   = (char*)d_ws;

    Params P;
    for (int sc = 0; sc < 2; ++sc) {
        int base = 2 + sc * 12;
        P.bn_cin_g[sc] = (const float*)d_in[base + 0];
        P.bn_cin_b[sc] = (const float*)d_in[base + 1];
        P.bn_cin_m[sc] = (const float*)d_in[base + 2];
        P.bn_cin_v[sc] = (const float*)d_in[base + 3];
        P.w_phi[sc]    = (const float*)d_in[base + 4];
        P.b_phi[sc]    = (const float*)d_in[base + 5];
        P.bn_phi_g[sc] = (const float*)d_in[base + 6];
        P.bn_phi_b[sc] = (const float*)d_in[base + 7];
        P.bn_phi_m[sc] = (const float*)d_in[base + 8];
        P.bn_phi_v[sc] = (const float*)d_in[base + 9];
        P.w_psi[sc]    = (const float*)d_in[base + 10];
        P.b_psi[sc]    = (const float*)d_in[base + 11];
    }

    int* idx0 = (int*)(ws + OFF_IDX0);
    int* idx1 = (int*)(ws + OFF_IDX1);

    int init_total = 3 * BN_PTS + BATCH * CFEAT * N_PTS;
    k_init_out<<<(init_total + 255) / 256, 256, 0, stream>>>(xyz, feat, dout);
    k_prep_scalars<<<1, 896, 0, stream>>>(P, ws);
    k_prep_Aphi<<<(2 * 16 * 5 * 32 + 255) / 256, 256, 0, stream>>>(P, ws);
    k_prep_Apsi<<<(2 * 4 * 8 * 32 + 255) / 256, 256, 0, stream>>>(P, ws);
    k_ballquery<<<BN_PTS / 4, 128, 0, stream>>>(xyz, idx0, idx1);

    size_t phi_lds = APHI_BYTES_PER_SC + 8 * C_OUT * sizeof(float);  // 90112 B
    k_phi<<<BN_PTS / 8, 256, phi_lds, stream>>>(xyz, feat, ws);
    k_psi<<<(BN_PTS / 16) / 4, 128, 0, stream>>>(ws, dout);
}
